// SingleHeadSelfAttention_65755949302229
// MI455X (gfx1250) — compile-verified, matrix operand order checked by reading, not run on hardware
//
#include <hip/hip_runtime.h>
#include <stdint.h>

#define B_DIM 8
#define S_DIM 2048
#define H_DIM 768

typedef __attribute__((ext_vector_type(16))) __bf16 v16bf;
typedef __attribute__((ext_vector_type(8)))  float  v8f;

union FragU { uint4 u[2]; v16bf v; };

// ---------------------------------------------------------------- helpers ---

__device__ __forceinline__ uint16_t f32_to_bf16(float f) {
  union { float f; uint32_t u; } cv; cv.f = f;
  uint32_t u = cv.u;
  if ((u & 0x7F800000u) == 0x7F800000u) return (uint16_t)(u >> 16); // Inf/NaN
  return (uint16_t)((u + 0x7FFFu + ((u >> 16) & 1u)) >> 16);        // RNE
}

// Load one 16-bit A/B WMMA fragment from a precomputed lane pointer
// p = base + row_offset + 8*hh (hh = lane/16). With k0 a compile-time
// constant (fully unrolled K loops) both chunks use immediate offsets.
//   elems 0..7  <- p[k0 .. k0+7]    (K = 8*hh + 0..7)
//   elems 8..15 <- p[k0+16.. k0+23] (K = 16 + 8*hh + 0..7)
__device__ __forceinline__ v16bf load_frag(const uint16_t* p, int k0) {
  FragU f;
  f.u[0] = *(const uint4*)(p + k0);
  f.u[1] = *(const uint4*)(p + k0 + 16);
  return f.v;
}

__device__ __forceinline__ v8f wmma_bf16(v16bf a, v16bf b, v8f c) {
  return __builtin_amdgcn_wmma_f32_16x16x32_bf16(
      /*neg_a=*/false, a, /*neg_b=*/false, b,
      /*c_mod=*/(short)0, c, /*reuse_a=*/false, /*reuse_b=*/false);
}

// --------------------------------------------------------------- kernels ----

// 4-wide f32 -> bf16 conversion (n must be a multiple of 4).
__global__ void cvt_f32_bf16_kernel(const float* __restrict__ in,
                                    uint16_t* __restrict__ out, int n4) {
  int i = blockIdx.x * blockDim.x + threadIdx.x;
  if (i < n4) {
    const float4 f = ((const float4*)in)[i];
    uint2 p;
    p.x = (uint32_t)f32_to_bf16(f.x) | ((uint32_t)f32_to_bf16(f.y) << 16);
    p.y = (uint32_t)f32_to_bf16(f.z) | ((uint32_t)f32_to_bf16(f.w) << 16);
    ((uint2*)out)[i] = p;
  }
}

// Q/K/V projection, 32M x 16N per wave: 6 WMMAs per K-step from 5 fragment
// loads, x-fragments shared by Q/K/V. K-loop fully unrolled, loop-invariant
// lane pointers -> loads use immediate offsets only.
// q,k stored row-major bf16 [B,S,H]; v stored transposed bf16 [B,H,S].
__global__ void qkv_kernel(const uint16_t* __restrict__ xb,
                           const uint16_t* __restrict__ wq,
                           const uint16_t* __restrict__ wk,
                           const uint16_t* __restrict__ wv,
                           uint16_t* __restrict__ qb,
                           uint16_t* __restrict__ kb,
                           uint16_t* __restrict__ vtb) {
  const int mt = blockIdx.x;            // 32-row tile (S/32)
  const int nt = blockIdx.y;            // 16-dim tile (H/16)
  const int b  = blockIdx.z;
  const int lane = threadIdx.x & 31;
  const int hh = lane >> 4;
  const int mn = lane & 15;

  const uint16_t* xp0 = xb + ((size_t)b * S_DIM + mt * 32 + mn) * H_DIM + 8 * hh;
  const uint16_t* xp1 = xp0 + 16 * H_DIM;
  const uint16_t* wqp = wq + (size_t)(nt * 16 + mn) * H_DIM + 8 * hh;
  const uint16_t* wkp = wk + (size_t)(nt * 16 + mn) * H_DIM + 8 * hh;
  const uint16_t* wvp = wv + (size_t)(nt * 16 + mn) * H_DIM + 8 * hh;

  v8f cq[2], ck[2], cv[2];
#pragma unroll
  for (int i = 0; i < 2; ++i) { cq[i] = {}; ck[i] = {}; cv[i] = {}; }

#pragma unroll
  for (int k0 = 0; k0 < H_DIM; k0 += 32) {
    v16bf a0 = load_frag(xp0, k0);
    v16bf a1 = load_frag(xp1, k0);
    v16bf bq = load_frag(wqp, k0);
    v16bf bk = load_frag(wkp, k0);
    v16bf bv = load_frag(wvp, k0);
    cq[0] = wmma_bf16(a0, bq, cq[0]);
    cq[1] = wmma_bf16(a1, bq, cq[1]);
    ck[0] = wmma_bf16(a0, bk, ck[0]);
    ck[1] = wmma_bf16(a1, bk, ck[1]);
    cv[0] = wmma_bf16(a0, bv, cv[0]);
    cv[1] = wmma_bf16(a1, bv, cv[1]);
  }

#pragma unroll
  for (int i = 0; i < 2; ++i) {
#pragma unroll
    for (int r = 0; r < 8; ++r) {
      int m = i * 16 + r + 8 * hh;
      size_t off = ((size_t)b * S_DIM + mt * 32 + m) * H_DIM + nt * 16 + mn;
      qb[off] = f32_to_bf16(cq[i][r]);
      kb[off] = f32_to_bf16(ck[i][r]);
    }
    // vT packed store: row d = nt*16+mn, 8 consecutive s positions.
    uint32_t pk[4];
#pragma unroll
    for (int r = 0; r < 4; ++r)
      pk[r] = (uint32_t)f32_to_bf16(cv[i][2 * r]) |
              ((uint32_t)f32_to_bf16(cv[i][2 * r + 1]) << 16);
    size_t voff = ((size_t)b * H_DIM + nt * 16 + mn) * S_DIM +
                  mt * 32 + i * 16 + 8 * hh;
    *(uint4*)(vtb + voff) = make_uint4(pk[0], pk[1], pk[2], pk[3]);
  }
}

// scores = scale * Q.K^T with key mask, softmax stats. One wave per
// (batch, 16-query tile); keys processed 64 at a time (4 WMMAs per K-step,
// A fragment from LDS via immediate-offset ds_load_b128).
__global__ void scores_softmax_kernel(const uint16_t* __restrict__ qb,
                                      const uint16_t* __restrict__ kb,
                                      const int* __restrict__ mask,
                                      float* __restrict__ scores,
                                      uint16_t* __restrict__ probs,
                                      float* __restrict__ rowscale) {
  __shared__ uint16_t qs[16 * H_DIM];   // 24 KB Q tile
  __shared__ float rmax_s[16];
  const int qt = blockIdx.x;
  const int b  = blockIdx.y;
  const int lane = threadIdx.x & 31;
  const int hh = lane >> 4;
  const int mn = lane & 15;
  const float sc = 0.03608439182435161f; // 1/sqrt(768)

  // Cooperative copy of the contiguous 16x768 Q tile into LDS (b128 chunks).
  {
    const uint4* src = (const uint4*)(qb + ((size_t)b * S_DIM + qt * 16) * H_DIM);
    uint4* dst = (uint4*)qs;
    for (int i = lane; i < (16 * H_DIM) / 8; i += 32) dst[i] = src[i];
  }
  __syncthreads();

  float rm[8];
#pragma unroll
  for (int r = 0; r < 8; ++r) rm[r] = -1e30f;

  float* stile = scores + ((size_t)b * S_DIM + qt * 16) * S_DIM;
  const uint16_t* kbase = kb + (size_t)b * S_DIM * H_DIM;
  const uint16_t* qsp = qs + mn * H_DIM + 8 * hh;   // LDS lane pointer
  const int nkt = S_DIM / 64;

  for (int kt = 0; kt < nkt; ++kt) {
    const uint16_t* kp0 = kbase + (size_t)(kt * 64 + mn) * H_DIM + 8 * hh;
    const uint16_t* kp1 = kp0 + 16 * H_DIM;
    const uint16_t* kp2 = kp0 + 32 * H_DIM;
    const uint16_t* kp3 = kp0 + 48 * H_DIM;
    if (kt + 1 < nkt) {   // prefetch next key tile rows (global_prefetch_b8)
      __builtin_prefetch(kp0 + 64 * H_DIM, 0, 1);
      __builtin_prefetch(kp1 + 64 * H_DIM, 0, 1);
      __builtin_prefetch(kp2 + 64 * H_DIM, 0, 1);
      __builtin_prefetch(kp3 + 64 * H_DIM, 0, 1);
    }

    v8f c[4];
#pragma unroll
    for (int n = 0; n < 4; ++n) c[n] = {};

#pragma unroll
    for (int k0 = 0; k0 < H_DIM; k0 += 32) {
      v16bf a = load_frag(qsp, k0);                 // LDS, immediate offsets
      c[0] = wmma_bf16(a, load_frag(kp0, k0), c[0]);
      c[1] = wmma_bf16(a, load_frag(kp1, k0), c[1]);
      c[2] = wmma_bf16(a, load_frag(kp2, k0), c[2]);
      c[3] = wmma_bf16(a, load_frag(kp3, k0), c[3]);
    }

#pragma unroll
    for (int n = 0; n < 4; ++n) {
      const int mk = mask[b * S_DIM + kt * 64 + n * 16 + mn];  // per-key=per-lane
#pragma unroll
      for (int r = 0; r < 8; ++r) {
        float s = mk ? (c[n][r] * sc) : -1e30f;
        rm[r] = fmaxf(rm[r], s);       // per-lane running max (no shuffles here)
        stile[(size_t)(r + 8 * hh) * S_DIM + kt * 64 + n * 16 + mn] = s;
      }
    }
  }

  // One cross-lane max reduction per row, broadcast via LDS.
#pragma unroll
  for (int r = 0; r < 8; ++r) {
    float v = rm[r];
    v = fmaxf(v, __shfl_xor(v, 1, 32));
    v = fmaxf(v, __shfl_xor(v, 2, 32));
    v = fmaxf(v, __shfl_xor(v, 4, 32));
    v = fmaxf(v, __shfl_xor(v, 8, 32));
    if (mn == 0) rmax_s[r + 8 * hh] = v;
  }
  __syncthreads();

  // Pass 2 (vectorized): lane owns half of row `mn` as alternating 4-column chunks.
  const float rmax = rmax_s[mn];
  const float* srow = stile + (size_t)mn * S_DIM;
  uint16_t* prow = probs + ((size_t)b * S_DIM + qt * 16 + mn) * S_DIM;
  float sum = 0.0f;
  for (int j = 0; j < S_DIM / 8; ++j) {
    const int col = 8 * j + 4 * hh;
    float4 s4 = *(const float4*)(srow + col);
    float p0 = __expf(s4.x - rmax);
    float p1 = __expf(s4.y - rmax);
    float p2 = __expf(s4.z - rmax);
    float p3 = __expf(s4.w - rmax);
    sum += (p0 + p1) + (p2 + p3);
    uint2 pk;
    pk.x = (uint32_t)f32_to_bf16(p0) | ((uint32_t)f32_to_bf16(p1) << 16);
    pk.y = (uint32_t)f32_to_bf16(p2) | ((uint32_t)f32_to_bf16(p3) << 16);
    *(uint2*)(prow + col) = pk;
  }
  sum += __shfl_xor(sum, 16, 32);      // combine the two half-rows
  if (hh == 0)
    rowscale[(size_t)b * S_DIM + qt * 16 + mn] = 1.0f / sum;
}

// output = rowscale * (P . V), 32M x 48N per wave: 6 WMMAs per K-step from
// 5 fragment loads; fully unrolled K with immediate-offset loads.
__global__ void av_kernel(const uint16_t* __restrict__ probs,
                          const uint16_t* __restrict__ vtb,
                          const float* __restrict__ rowscale,
                          float* __restrict__ out) {
  const int qt = blockIdx.x;            // S/32
  const int nt = blockIdx.y;            // H/48
  const int b  = blockIdx.z;
  const int lane = threadIdx.x & 31;
  const int hh = lane >> 4;
  const int mn = lane & 15;

  const uint16_t* pp0 = probs + ((size_t)b * S_DIM + qt * 32 + mn) * S_DIM + 8 * hh;
  const uint16_t* pp1 = pp0 + 16 * S_DIM;
  const uint16_t* vp0 = vtb + ((size_t)b * H_DIM + nt * 48 + mn) * S_DIM + 8 * hh;
  const uint16_t* vp1 = vp0 + 16 * S_DIM;
  const uint16_t* vp2 = vp0 + 32 * S_DIM;

  v8f c[2][3];
#pragma unroll
  for (int i = 0; i < 2; ++i)
#pragma unroll
    for (int j = 0; j < 3; ++j) c[i][j] = {};

#pragma unroll
  for (int k0 = 0; k0 < S_DIM; k0 += 32) {
    v16bf a0 = load_frag(pp0, k0);
    v16bf a1 = load_frag(pp1, k0);
    v16bf b0 = load_frag(vp0, k0);
    v16bf b1 = load_frag(vp1, k0);
    v16bf b2 = load_frag(vp2, k0);
    c[0][0] = wmma_bf16(a0, b0, c[0][0]);
    c[1][0] = wmma_bf16(a1, b0, c[1][0]);
    c[0][1] = wmma_bf16(a0, b1, c[0][1]);
    c[1][1] = wmma_bf16(a1, b1, c[1][1]);
    c[0][2] = wmma_bf16(a0, b2, c[0][2]);
    c[1][2] = wmma_bf16(a1, b2, c[1][2]);
  }

#pragma unroll
  for (int i = 0; i < 2; ++i)
#pragma unroll
    for (int r = 0; r < 8; ++r) {
      const int m = i * 16 + r + 8 * hh;
      const float scl = rowscale[(size_t)b * S_DIM + qt * 32 + m];
      float* orow = out + ((size_t)b * S_DIM + qt * 32 + m) * H_DIM + nt * 48;
#pragma unroll
      for (int j = 0; j < 3; ++j)
        orow[j * 16 + mn] = c[i][j][r] * scl;
    }
}

// ---------------------------------------------------------------- launch ----

extern "C" void kernel_launch(void* const* d_in, const int* in_sizes, int n_in,
                              void* d_out, int out_size, void* d_ws, size_t ws_size,
                              hipStream_t stream) {
  const float* x    = (const float*)d_in[0];
  const int*   mask = (const int*)d_in[1];
  const float* Wq   = (const float*)d_in[2];
  const float* Wk   = (const float*)d_in[3];
  const float* Wv   = (const float*)d_in[4];
  float* out = (float*)d_out;

  const size_t nx = (size_t)B_DIM * S_DIM * H_DIM;   // 12.58M
  const size_t nw = (size_t)H_DIM * H_DIM;           // 589k
  const size_t ns = (size_t)B_DIM * S_DIM * S_DIM;   // 33.5M

  char* ws = (char*)d_ws;
  size_t off = 0;
  auto carve = [&](size_t bytes) {
    void* p = ws + off;
    off = (off + bytes + 255) & ~(size_t)255;
    return p;
  };
  uint16_t* xb   = (uint16_t*)carve(nx * 2);
  uint16_t* wqb  = (uint16_t*)carve(nw * 2);
  uint16_t* wkb  = (uint16_t*)carve(nw * 2);
  uint16_t* wvb  = (uint16_t*)carve(nw * 2);
  uint16_t* qb   = (uint16_t*)carve(nx * 2);
  uint16_t* kb   = (uint16_t*)carve(nx * 2);
  uint16_t* vtb  = (uint16_t*)carve(nx * 2);
  float*    scf  = (float*)carve(ns * 4);
  uint16_t* prb  = (uint16_t*)carve(ns * 2);
  float*    rsc  = (float*)carve((size_t)B_DIM * S_DIM * 4);
  (void)ws_size; (void)in_sizes; (void)n_in; (void)out_size;

  cvt_f32_bf16_kernel<<<(unsigned)((nx / 4 + 255) / 256), 256, 0, stream>>>(x, xb, (int)(nx / 4));
  cvt_f32_bf16_kernel<<<(unsigned)((nw / 4 + 255) / 256), 256, 0, stream>>>(Wq, wqb, (int)(nw / 4));
  cvt_f32_bf16_kernel<<<(unsigned)((nw / 4 + 255) / 256), 256, 0, stream>>>(Wk, wkb, (int)(nw / 4));
  cvt_f32_bf16_kernel<<<(unsigned)((nw / 4 + 255) / 256), 256, 0, stream>>>(Wv, wvb, (int)(nw / 4));

  qkv_kernel<<<dim3(S_DIM / 32, H_DIM / 16, B_DIM), 32, 0, stream>>>(
      xb, wqb, wkb, wvb, qb, kb, vtb);

  scores_softmax_kernel<<<dim3(S_DIM / 16, B_DIM), 32, 0, stream>>>(
      qb, kb, mask, scf, prb, rsc);

  av_kernel<<<dim3(S_DIM / 32, H_DIM / 48, B_DIM), 32, 0, stream>>>(
      prb, vtb, rsc, out);
}
